// Attention_6322191859738
// MI455X (gfx1250) — matrix operand order repaired and checked by reading, not compiled, not run on hardware
//
#include <hip/hip_runtime.h>
#include <stdint.h>

typedef _Float16 f16;
typedef __attribute__((ext_vector_type(16))) _Float16 v16h;
typedef __attribute__((ext_vector_type(8)))  float    v8f;

#define HEADS 4
#define DHEAD 32
#define SEQ   4096      // 16*16*16 spatial positions
#define CIN   64
#define HID   128       // HEADS*DHEAD

union HF { v16h v; uint32_t u[8]; };

// Raw max: avoids fmaxf's operand canonicalization (v_max v,x,x pairs).
__device__ __forceinline__ float fmax_raw(float a, float b) {
    float r;
    asm("v_max_num_f32_e32 %0, %1, %2" : "=v"(r) : "v"(a), "v"(b));
    return r;
}

// Fused DPP16 butterfly reductions over the 16 lanes of a lane-half:
// one VALU instruction per step (DPP as src0 modifier).
#define DPP_MAX_STEP(x, CTRL)                                                  \
    asm("v_max_num_f32_dpp %0, %0, %0 " CTRL                                   \
        " row_mask:0xf bank_mask:0xf bound_ctrl:1" : "+v"(x))
#define DPP_ADD_STEP(x, CTRL)                                                  \
    asm("v_add_f32_dpp %0, %0, %0 " CTRL                                       \
        " row_mask:0xf bank_mask:0xf bound_ctrl:1" : "+v"(x))

__device__ __forceinline__ float row_max16(float x) {
    DPP_MAX_STEP(x, "quad_perm:[1,0,3,2]");   // xor1
    DPP_MAX_STEP(x, "quad_perm:[2,3,0,1]");   // xor2
    DPP_MAX_STEP(x, "row_half_mirror");       // quads uniform -> xor4
    DPP_MAX_STEP(x, "row_mirror");            // octets uniform -> xor8
    return x;
}
__device__ __forceinline__ float row_sum16(float x) {
    DPP_ADD_STEP(x, "quad_perm:[1,0,3,2]");
    DPP_ADD_STEP(x, "quad_perm:[2,3,0,1]");
    DPP_ADD_STEP(x, "row_half_mirror");
    DPP_ADD_STEP(x, "row_mirror");
    return x;
}

// ---------------------------------------------------------------------------
// Kernel 1: QKV projection (1x1 conv == channel matmul), fp32 in, f16 out.
// q pre-scaled by dim_head^-0.5 * log2(e), stored [h][n][32] (A layout).
// k pair-packed  [h][d/2][n]{d,d+1}  -> B-frag dword = 1 coalesced b32 load.
// v pair-packed  [h][n/2][32]{n,n+1} -> B-frag dword = 1 coalesced b32 load.
// ---------------------------------------------------------------------------
__global__ __launch_bounds__(256)
void qkv_kernel(const float* __restrict__ x, const float* __restrict__ wqkv,
                f16* __restrict__ qt, f16* __restrict__ kp, f16* __restrict__ vp) {
    int idx = blockIdx.x * 256 + threadIdx.x;
    if (idx >= 3 * HID * SEQ) return;
    int n = idx & (SEQ - 1);
    int o = idx >> 12;                       // / SEQ
    const float* wr = wqkv + o * CIN;
    const float* xc = x + n;
    float acc = 0.f;
#pragma unroll 8
    for (int c = 0; c < CIN; ++c)
        acc = fmaf(wr[c], xc[(size_t)c * SEQ], acc);
    if (o < HID) {                           // Q (scale * log2e folded in)
        int h = o >> 5, d = o & 31;
        qt[((size_t)h * SEQ + n) * DHEAD + d] = (f16)(acc * 0.25504672320321864f);
    } else if (o < 2 * HID) {                // K packed pairs along d
        int oo = o - HID, h = oo >> 5, d = oo & 31;
        kp[(((size_t)h * 16 + (d >> 1)) * SEQ + n) * 2 + (d & 1)] = (f16)acc;
    } else {                                 // V packed pairs along n
        int oo = o - 2 * HID, h = oo >> 5, d = oo & 31;
        vp[(((size_t)h * (SEQ / 2) + (n >> 1)) * DHEAD + d) * 2 + (n & 1)] = (f16)acc;
    }
}

// ---------------------------------------------------------------------------
// Kernel 2: flash attention, one wave per 16-row query tile.
// Per 32-wide j chunk: 2x S-WMMA (K = full head dim) -> online softmax on the
// f32 C fragments (fused-DPP max reduction, raw v_exp_f32, deferred sum) ->
// P through LDS into A-fragment layout -> 2x PV-WMMA accumulating O.
// ---------------------------------------------------------------------------
__global__ __launch_bounds__(128)
void attn_kernel(const f16* __restrict__ qt, const f16* __restrict__ kp,
                 const f16* __restrict__ vp, float* __restrict__ obuf) {
    __shared__ f16 lds_p[4][16][32];         // per-wave 16x32 P tile, row-major

    const int lane = threadIdx.x & 31;
    const int wave = threadIdx.x >> 5;
    const int bph  = SEQ / 64;               // blocks per head
    const int h    = blockIdx.x / bph;
    const int i0   = (blockIdx.x % bph) * 64 + wave * 16;

    const int g = lane >> 4;                 // lane half
    const int m = lane & 15;                 // row (A) / col (B,C) selector

    // A-fragment dword indices (K-pair index) per VGPR v, contiguous k order (k = 16g + 2v, +1)
    int kd[8];
#pragma unroll
    for (int v = 0; v < 8; ++v)
        kd[v] = v + 8 * g;

    // ---- Q A-fragment: rows i0..i0+15, K = d = 0..31 (2x b128-able) ----
    HF aq;
    const uint32_t* q32 = (const uint32_t*)qt + ((size_t)h * SEQ + (i0 + m)) * 16;
#pragma unroll
    for (int v = 0; v < 8; ++v) aq.u[v] = q32[kd[v]];

    v8f oacc0 = {}, oacc1 = {};
    float mrow[8], lpart[8];                 // lpart: per-LANE partial row sums
#pragma unroll
    for (int v = 0; v < 8; ++v) { mrow[v] = -1e30f; lpart[v] = 0.f; }

    const uint32_t* k32 = (const uint32_t*)kp + ((size_t)h * 16 + 8 * g) * SEQ + m;
    const uint32_t* v32 = (const uint32_t*)vp + ((size_t)h * (SEQ / 2) + 8 * g) * DHEAD + m;

    for (int j0 = 0; j0 < SEQ; j0 += 32) {
        // ---- K B-fragments: dword v = K-pair (16g+2v, +1), N = j0+m (+16) ----
        HF bk0, bk1;
#pragma unroll
        for (int v = 0; v < 8; ++v) {
            const uint32_t* p = k32 + (size_t)v * SEQ + j0;
            bk0.u[v] = p[0];
            bk1.u[v] = p[16];
        }
        if (j0 + 32 < SEQ) {                 // prefetch next K/V chunk
            __builtin_prefetch(k32 + j0 + 32, 0, 3);
            __builtin_prefetch(v32 + ((size_t)(j0 + 32) >> 1) * DHEAD, 0, 3);
        }
        v8f s0 = {}, s1 = {};
        s0 = __builtin_amdgcn_wmma_f32_16x16x32_f16(false, aq.v, false, bk0.v,
                                                    (short)0, s0, false, false);
        s1 = __builtin_amdgcn_wmma_f32_16x16x32_f16(false, aq.v, false, bk1.v,
                                                    (short)0, s1, false, false);

        // ---- online softmax (base-2 domain); row r = v + 8g per lane-half ----
#pragma unroll
        for (int v = 0; v < 8; ++v) {
            float cmax  = row_max16(fmax_raw(s0[v], s1[v]));
            float mnew  = fmax_raw(mrow[v], cmax);
            float scale = __builtin_amdgcn_exp2f(mrow[v] - mnew); // row-uniform
            float p0 = __builtin_amdgcn_exp2f(s0[v] - mnew);
            float p1 = __builtin_amdgcn_exp2f(s1[v] - mnew);
            lpart[v] = fmaf(lpart[v], scale, p0 + p1);            // lane-local
            mrow[v] = mnew;
            oacc0[v] *= scale;
            oacc1[v] *= scale;
            int r = v + 8 * g;
            lds_p[wave][r][m]      = (f16)p0;
            lds_p[wave][r][16 + m] = (f16)p1;
        }
        asm volatile("s_wait_dscnt 0" ::: "memory");   // P stores -> A loads

        // ---- P A-fragment from LDS (dword loads, 2x b128-able) ----
        HF ap;
        const uint32_t* lp = (const uint32_t*)&lds_p[wave][m][0];
#pragma unroll
        for (int v = 0; v < 8; ++v) ap.u[v] = lp[kd[v]];

        // ---- V B-fragments: dword v = K-pair (j0+16g+2v, +1), N = d ----
        HF bv0, bv1;
#pragma unroll
        for (int v = 0; v < 8; ++v) {
            const uint32_t* p = v32 + ((size_t)(j0 >> 1) + v) * DHEAD;
            bv0.u[v] = p[0];
            bv1.u[v] = p[16];
        }
        oacc0 = __builtin_amdgcn_wmma_f32_16x16x32_f16(false, ap.v, false, bv0.v,
                                                       (short)0, oacc0, false, false);
        oacc1 = __builtin_amdgcn_wmma_f32_16x16x32_f16(false, ap.v, false, bv1.v,
                                                       (short)0, oacc1, false, false);
        asm volatile("s_wait_dscnt 0" ::: "memory");   // A loads done before next stores
    }

    // ---- deferred row-sum reduction, normalize, write O [c = h*32+d][n] ----
#pragma unroll
    for (int v = 0; v < 8; ++v) {
        float lrow = row_sum16(lpart[v]);
        float inv  = __builtin_amdgcn_rcpf(lrow);
        int r = v + 8 * g;
        int i = i0 + r;
        obuf[(size_t)(h * DHEAD + m) * SEQ + i]      = oacc0[v] * inv;
        obuf[(size_t)(h * DHEAD + 16 + m) * SEQ + i] = oacc1[v] * inv;
    }
}

// ---------------------------------------------------------------------------
// Kernel 3: output projection y = w_out @ O + b_out
// ---------------------------------------------------------------------------
__global__ __launch_bounds__(256)
void proj_kernel(const float* __restrict__ obuf, const float* __restrict__ wout,
                 const float* __restrict__ bout, float* __restrict__ y) {
    int idx = blockIdx.x * 256 + threadIdx.x;
    if (idx >= CIN * SEQ) return;
    int n = idx & (SEQ - 1);
    int o = idx >> 12;
    const float* wr = wout + o * HID;
    float acc = bout[o];
#pragma unroll 8
    for (int c = 0; c < HID; ++c)
        acc = fmaf(wr[c], obuf[(size_t)c * SEQ + n], acc);
    y[idx] = acc;
}

extern "C" void kernel_launch(void* const* d_in, const int* in_sizes, int n_in,
                              void* d_out, int out_size, void* d_ws, size_t ws_size,
                              hipStream_t stream) {
    const float* x     = (const float*)d_in[0];   // [1,64,16,16,16]
    const float* wqkv  = (const float*)d_in[1];   // [384,64]
    const float* wout  = (const float*)d_in[2];   // [64,128]
    const float* bout  = (const float*)d_in[3];   // [64]
    float* y = (float*)d_out;                     // [1,64,16,16,16]

    const size_t QT = (size_t)HEADS * SEQ * DHEAD * sizeof(f16);  // 1 MB each
    char* ws = (char*)d_ws;
    f16*   qt   = (f16*)(ws);
    f16*   kp   = (f16*)(ws + QT);
    f16*   vp   = (f16*)(ws + 2 * QT);
    float* obuf = (float*)(ws + 3 * QT);          // [128][4096] f32, 2 MB

    qkv_kernel <<<(3 * HID * SEQ + 255) / 256, 256, 0, stream>>>(x, wqkv, qt, kp, vp);
    attn_kernel<<<HEADS * (SEQ / 64), 128, 0, stream>>>(qt, kp, vp, obuf);
    proj_kernel<<<(CIN * SEQ + 255) / 256, 256, 0, stream>>>(obuf, wout, bout, y);
}
